// CenterLoss_64080912056737
// MI455X (gfx1250) — compile-verified, matrix operand order checked by reading, not run on hardware
//
#include <hip/hip_runtime.h>
#include <hip/hip_bf16.h>

#define NUM_CLASSES 21
#define FEATSZ      19
#define N_ANCH      9
#define NB          256
#define NUM_OBJS    12
#define PRI         (FEATSZ*FEATSZ*N_ANCH)   /* 3249 */
#define CELLS       (FEATSZ*FEATSZ)          /* 361  */
#define THRESHOLD   0.5f
#define NEGPOS      3
#define VAR0        0.1f
#define VAR1        0.2f
#define CHUNK       ((PRI + 255) / 256)      /* 13 */

typedef __attribute__((ext_vector_type(16))) _Float16 v16h;
typedef __attribute__((ext_vector_type(8)))  float    v8f;

// ---------------------------------------------------------------------------
// Kernel 1: per-batch matching + smooth-L1 + per-prior cross entropy.
// One workgroup (256 threads = 8 wave32) per batch element.
// ---------------------------------------------------------------------------
__global__ __launch_bounds__(256) void ssd_match_ce_kernel(
    const float* __restrict__ loc_data,   // [B,P,4]
    const float* __restrict__ conf_data,  // [B,P,21]
    const float* __restrict__ priors,     // [P,4] (cx,cy,w,h)
    const float* __restrict__ targets,    // [B,12,5] (x1,y1,x2,y2,label)
    float* __restrict__ ws_ce,            // [B,P]
    int*   __restrict__ ws_conf,          // [B,P]
    float* __restrict__ ws_lossl,         // [B]
    int*   __restrict__ ws_npos)          // [B]
{
    const int b   = blockIdx.x;
    const int tid = threadIdx.x;

    __shared__ float s_truth[NUM_OBJS][4];
    __shared__ float s_area[NUM_OBJS];
    __shared__ int   s_label[NUM_OBJS];
    __shared__ unsigned long long s_bpkey[NUM_OBJS];  // packed (ov_bits<<32)|(~prior)
    __shared__ float s_ov[PRI];
    __shared__ int   s_ti[PRI];
    __shared__ float s_redf[256];
    __shared__ int   s_redi[256];

    if (tid < NUM_OBJS) {
        const float* t = targets + ((size_t)b * NUM_OBJS + tid) * 5;
        s_truth[tid][0] = t[0]; s_truth[tid][1] = t[1];
        s_truth[tid][2] = t[2]; s_truth[tid][3] = t[3];
        s_area[tid]  = (t[2] - t[0]) * (t[3] - t[1]);
        s_label[tid] = (int)t[4];
        s_bpkey[tid] = 0ull;
    }
    __syncthreads();

    // ---- IoU: per-prior best truth (first-max tiebreak) and per-truth best prior ----
    unsigned long long lkey[NUM_OBJS];
    #pragma unroll
    for (int t = 0; t < NUM_OBJS; ++t) lkey[t] = 0ull;

    for (int p = tid; p < PRI; p += 256) {
        const float cx = priors[p*4+0], cy = priors[p*4+1];
        const float pw = priors[p*4+2], ph = priors[p*4+3];
        const float px1 = cx - 0.5f*pw, py1 = cy - 0.5f*ph;
        const float px2 = cx + 0.5f*pw, py2 = cy + 0.5f*ph;
        const float parea = (px2 - px1) * (py2 - py1);
        float bov = -1.0f; int bt = 0;
        #pragma unroll
        for (int t = 0; t < NUM_OBJS; ++t) {
            const float ix1 = fmaxf(px1, s_truth[t][0]);
            const float iy1 = fmaxf(py1, s_truth[t][1]);
            const float ix2 = fminf(px2, s_truth[t][2]);
            const float iy2 = fminf(py2, s_truth[t][3]);
            const float inter = fmaxf(ix2 - ix1, 0.0f) * fmaxf(iy2 - iy1, 0.0f);
            const float ov = inter / (s_area[t] + parea - inter);
            if (ov > bov) { bov = ov; bt = t; }          // first max wins (jnp.argmax axis=0)
            const unsigned long long key =
                ((unsigned long long)__float_as_uint(ov) << 32) |
                (unsigned long long)(0xFFFFFFFFu - (unsigned)p); // max ov, tie -> min prior
            if (key > lkey[t]) lkey[t] = key;
        }
        s_ov[p] = bov; s_ti[p] = bt;
    }
    #pragma unroll
    for (int t = 0; t < NUM_OBJS; ++t) atomicMax(&s_bpkey[t], lkey[t]);
    __syncthreads();

    unsigned bp[NUM_OBJS];
    #pragma unroll
    for (int t = 0; t < NUM_OBJS; ++t)
        bp[t] = 0xFFFFFFFFu - (unsigned)(s_bpkey[t] & 0xFFFFFFFFull);

    // ---- force-match, encode, smooth-L1, cross-entropy ----
    float lossl = 0.0f; int npos = 0;
    for (int p = tid; p < PRI; p += 256) {
        float ov = s_ov[p]; int ti = s_ti[p];
        #pragma unroll
        for (int t = 0; t < NUM_OBJS; ++t)            // ascending: last writer wins
            if (bp[t] == (unsigned)p) { ov = 2.0f; ti = t; }

        const int conf = (ov < THRESHOLD) ? 0 : (s_label[ti] + 1);

        const float cx = priors[p*4+0], cy = priors[p*4+1];
        const float pw = priors[p*4+2], ph = priors[p*4+3];
        const float x1 = s_truth[ti][0], y1 = s_truth[ti][1];
        const float x2 = s_truth[ti][2], y2 = s_truth[ti][3];
        float g[4];
        g[0] = (0.5f*(x1 + x2) - cx) / (VAR0 * pw);
        g[1] = (0.5f*(y1 + y2) - cy) / (VAR0 * ph);
        g[2] = logf((x2 - x1) / pw) / VAR1;
        g[3] = logf((y2 - y1) / ph) / VAR1;

        if (conf > 0) {
            const float* ld = loc_data + ((size_t)b * PRI + p) * 4;
            #pragma unroll
            for (int d = 0; d < 4; ++d) {
                const float df = ld[d] - g[d];
                const float ad = fabsf(df);
                lossl += (ad < 1.0f) ? 0.5f * df * df : (ad - 0.5f);
            }
            ++npos;
        }

        const float* cd = conf_data + ((size_t)b * PRI + p) * NUM_CLASSES;
        __builtin_prefetch(cd + 256 * NUM_CLASSES, 0, 1);   // stream-ahead: global_prefetch_b8
        float m = cd[0];
        #pragma unroll
        for (int c = 1; c < NUM_CLASSES; ++c) m = fmaxf(m, cd[c]);
        float s = 0.0f;
        #pragma unroll
        for (int c = 0; c < NUM_CLASSES; ++c) s += expf(cd[c] - m);
        const float ce = (m + logf(s)) - cd[conf];

        ws_ce  [(size_t)b * PRI + p] = ce;
        ws_conf[(size_t)b * PRI + p] = conf;
    }

    s_redf[tid] = lossl; s_redi[tid] = npos; __syncthreads();
    for (int off = 128; off > 0; off >>= 1) {
        if (tid < off) { s_redf[tid] += s_redf[tid+off]; s_redi[tid] += s_redi[tid+off]; }
        __syncthreads();
    }
    if (tid == 0) { ws_lossl[b] = s_redf[0]; ws_npos[b] = s_redi[0]; }
}

// ---------------------------------------------------------------------------
// Kernel 2: per-batch hard-negative mining (exact stable top-k via radix
// binary search on float bit patterns), loss_c partial, feature maps.
// The ce row is staged memory->LDS on the CDNA5 async path
// (global_load_async_to_lds_b128, ASYNCcnt), overlapped with the conf_t
// staging on the VGPR path, then fenced with s_wait_asynccnt 0.
// ---------------------------------------------------------------------------
__global__ __launch_bounds__(256) void ssd_mine_kernel(
    const float* __restrict__ ws_ce,
    const int*   __restrict__ ws_conf,
    const int*   __restrict__ ws_npos,
    float* __restrict__ ws_lossc,
    float* __restrict__ out_conf_fm,     // [B,361]
    float* __restrict__ out_have)        // [B,361]
{
    const int b = blockIdx.x, tid = threadIdx.x;
    __shared__ float          s_ce[PRI];
    __shared__ unsigned       s_bits[PRI];
    __shared__ unsigned char  s_conf[PRI];
    __shared__ unsigned char  s_sel[PRI];
    __shared__ int            s_scan[256];
    __shared__ float          s_redf[256];

    // ---- async stage: ce row (12996 B) -> LDS; 812 x b128 + 1 x b32 tail ----
    {
        const unsigned long long gbase =
            (unsigned long long)(ws_ce + (size_t)b * PRI);
        const unsigned lbase = (unsigned)(unsigned long long)&s_ce[0]; // LDS byte offset
        for (int i = tid; i < (PRI * 4) / 16; i += 256) {              // 812 chunks
            const unsigned long long ga = gbase + (unsigned long long)i * 16u;
            const unsigned la = lbase + (unsigned)i * 16u;
            asm volatile("global_load_async_to_lds_b128 %0, %1, off"
                         :: "v"(la), "v"(ga) : "memory");
        }
        if (tid == 0) {                                                // tail float 3248
            const unsigned long long ga = gbase + (unsigned long long)(PRI - 1) * 4u;
            const unsigned la = lbase + (unsigned)(PRI - 1) * 4u;
            asm volatile("global_load_async_to_lds_b32 %0, %1, off"
                         :: "v"(la), "v"(ga) : "memory");
        }
    }
    // overlapped with the async DMA: conf_t row via the VGPR path
    for (int p = tid; p < PRI; p += 256)
        s_conf[p] = (unsigned char)ws_conf[(size_t)b * PRI + p];

    asm volatile("s_wait_asynccnt 0x0" ::: "memory");
    __syncthreads();

    for (int p = tid; p < PRI; p += 256) {
        const float mine = (s_conf[p] > 0) ? 0.0f : fmaxf(s_ce[p], 0.0f); // ce >= 0
        s_bits[p] = __float_as_uint(mine);                                // monotone, v >= 0
    }
    __syncthreads();

    const int npos = ws_npos[b];
    const int k    = min(NEGPOS * npos, PRI - 1);

    const int lo = tid * CHUNK;
    const int hi = min(lo + CHUNK, PRI);

    unsigned tstar = 0; int r = 0;
    if (k > 0) {
        unsigned cur = 0;
        for (int bit = 30; bit >= 0; --bit) {
            const unsigned cand = cur | (1u << bit);
            int lc = 0;
            for (int p = lo; p < hi; ++p) lc += (s_bits[p] >= cand);
            s_scan[tid] = lc; __syncthreads();
            for (int off = 128; off > 0; off >>= 1) {
                if (tid < off) s_scan[tid] += s_scan[tid+off];
                __syncthreads();
            }
            const int cnt = s_scan[0]; __syncthreads();
            if (cnt >= k) cur = cand;                 // largest t with count(>=t) >= k
        }
        tstar = cur;
        int lc = 0;
        for (int p = lo; p < hi; ++p) lc += (s_bits[p] > tstar);
        s_scan[tid] = lc; __syncthreads();
        for (int off = 128; off > 0; off >>= 1) {
            if (tid < off) s_scan[tid] += s_scan[tid+off];
            __syncthreads();
        }
        r = k - s_scan[0];                            // ties to accept, in index order
        __syncthreads();
    }

    // stable tie ranking: exclusive prefix over tie counts (contiguous chunks)
    int lties = 0;
    for (int p = lo; p < hi; ++p) lties += (s_bits[p] == tstar);
    s_scan[tid] = lties; __syncthreads();
    for (int off = 1; off < 256; off <<= 1) {         // Hillis-Steele inclusive scan
        int v = s_scan[tid];
        if (tid >= off) v += s_scan[tid - off];
        __syncthreads(); s_scan[tid] = v; __syncthreads();
    }
    int base = s_scan[tid] - lties;                   // exclusive prefix

    float lossc = 0.0f;
    for (int p = lo; p < hi; ++p) {
        const bool pos = s_conf[p] > 0;
        bool neg = false;
        if (k > 0) {
            if (s_bits[p] > tstar) neg = true;
            else if (s_bits[p] == tstar && base < r) neg = true;
        }
        if (s_bits[p] == tstar) ++base;
        const bool sel = pos || neg;
        s_sel[p] = sel ? 1 : 0;
        if (sel) lossc += s_ce[p];
    }

    s_redf[tid] = lossc; __syncthreads();
    for (int off = 128; off > 0; off >>= 1) {
        if (tid < off) s_redf[tid] += s_redf[tid+off];
        __syncthreads();
    }
    if (tid == 0) ws_lossc[b] = s_redf[0];

    // feature maps: max over the 9 anchors of each 19x19 cell
    for (int c = tid; c < CELLS; c += 256) {
        int mc = 0, ms = 0;
        #pragma unroll
        for (int a = 0; a < N_ANCH; ++a) {
            const int p = c * N_ANCH + a;
            mc = max(mc, (int)s_conf[p]);
            ms = max(ms, (int)s_sel[p]);
        }
        out_conf_fm[(size_t)b * CELLS + c] = (float)mc;
        out_have  [(size_t)b * CELLS + c] = (float)ms;
    }
}

// ---------------------------------------------------------------------------
// Kernel 3: finalize. Single wave32. Each 256-element partial array is summed
// on the matrix pipe: f32 -> (hi,lo) f16 Dekker split packs 512 halves into
// one 16x32 A-tile, B = ones, f32 accumulation -> ~f32-accurate total.
// Emits v_wmma_f32_16x16x32_f16.
// ---------------------------------------------------------------------------
template <typename T>
__device__ inline float wmma_sum256(const T* __restrict__ src, int lane) {
    v16h bo;
    #pragma unroll
    for (int i = 0; i < 16; ++i) bo[i] = (_Float16)1.0f;
    v16h a;
    #pragma unroll
    for (int e = 0; e < 16; ++e) {
        const int idx = lane * 16 + e;        // 0..511 bijective over A slots
        const int s   = idx & 255;
        const float v = (float)src[s];
        const _Float16 h = (_Float16)v;
        a[e] = (idx < 256) ? h : (_Float16)(v - (float)h);   // hi then residual
    }
    v8f c = {};
    c = __builtin_amdgcn_wmma_f32_16x16x32_f16(false, a, false, bo,
                                               (short)0, c, false, false);
    float s = 0.0f;
    #pragma unroll
    for (int i = 0; i < 8; ++i) s += c[i];    // D[i,j] = rowsum_i, replicated over 16 cols
    #pragma unroll
    for (int off = 16; off > 0; off >>= 1) s += __shfl_xor(s, off, 32);
    return s * (1.0f / 16.0f);
}

__global__ __launch_bounds__(32) void ssd_finalize_kernel(
    const float* __restrict__ ws_lossl,
    const float* __restrict__ ws_lossc,
    const int*   __restrict__ ws_npos,
    float* __restrict__ d_out)
{
    const int lane = threadIdx.x;
    const float Sl = wmma_sum256(ws_lossl, lane);
    const float Sc = wmma_sum256(ws_lossc, lane);
    const float N  = wmma_sum256(ws_npos,  lane);
    if (lane == 0) {
        d_out[0] = Sl / N;
        d_out[1] = Sc / N;
    }
}

// ---------------------------------------------------------------------------
extern "C" void kernel_launch(void* const* d_in, const int* in_sizes, int n_in,
                              void* d_out, int out_size, void* d_ws, size_t ws_size,
                              hipStream_t stream) {
    const float* loc_data  = (const float*)d_in[0];
    const float* conf_data = (const float*)d_in[1];
    const float* priors    = (const float*)d_in[2];
    const float* targets   = (const float*)d_in[3];
    float* out = (float*)d_out;

    char* ws = (char*)d_ws;
    float* ws_ce    = (float*)ws;                                   // B*P f32
    int*   ws_conf  = (int*)(ws + sizeof(float) * (size_t)NB * PRI);// B*P i32
    float* ws_lossl = (float*)(ws + 2 * sizeof(float) * (size_t)NB * PRI);
    float* ws_lossc = ws_lossl + NB;
    int*   ws_npos  = (int*)(ws_lossc + NB);

    float* out_conf_fm = out + 2;
    float* out_have    = out + 2 + (size_t)NB * CELLS;

    ssd_match_ce_kernel<<<NB, 256, 0, stream>>>(loc_data, conf_data, priors, targets,
                                                ws_ce, ws_conf, ws_lossl, ws_npos);
    ssd_mine_kernel<<<NB, 256, 0, stream>>>(ws_ce, ws_conf, ws_npos,
                                            ws_lossc, out_conf_fm, out_have);
    ssd_finalize_kernel<<<1, 32, 0, stream>>>(ws_lossl, ws_lossc, ws_npos, out);
}
